// NeuralCDE_28707561406833
// MI455X (gfx1250) — compile-verified, matrix operand order checked by reading, not run on hardware
//
#include <hip/hip_runtime.h>

// ---------------------------------------------------------------------------
// Neural CDE (Heun, 100 steps) fully fused for gfx1250 (MI455X).
// Transposed formulation: H^T = W * Y^T per layer, so WEIGHTS are the WMMA A
// operand (pre-packed bf16 A-fragments in d_ws) and ACTIVATIONS are the B
// operand. The C-fragment output of one layer converts to the next layer's
// B-fragment purely in-lane (pack c[2r],c[2r+1] of two 16-row tiles), so the
// 100-step loop needs NO LDS staging and no cross-lane data movement except a
// single shfl_xor(16) in the final cd-contraction.
//   - one wave == one 16-column batch tile; y^T kept in C-fragments (regs)
//   - amdgpu_waves_per_eu(4) caps VGPRs at 256 (avoid s_set_vgpr_msb storm)
//   - per-step asm memory barrier stops LICM from hoisting+spilling the
//     loop-invariant weight loads to scratch
//   - relu = single v_max_num_f32 (inline asm; WMMA output is never sNaN)
//   - tanh = native v_tanh_f32 when the builtin exists (ISA lists TANH as a
//     CDNA5 transcendental), else exp-based 7-op fallback
// ---------------------------------------------------------------------------

typedef __attribute__((ext_vector_type(16))) __bf16        v16bf;
typedef __attribute__((ext_vector_type(2)))  __bf16        v2bf;
typedef __attribute__((ext_vector_type(8)))  float         v8f;
typedef __attribute__((ext_vector_type(4)))  unsigned int  u32x4;

union Frag { v16bf v; unsigned int u[8]; u32x4 q[2]; };

#define DEVFN static __device__ __forceinline__

DEVFN unsigned short f2bf(float x) {            // RNE, used by prep kernel
  union { float f; unsigned int u; } c; c.f = x;
  unsigned int u = c.u;
  unsigned int r = u + 0x7FFFu + ((u >> 16) & 1u);
  return (unsigned short)(r >> 16);
}

DEVFN unsigned int pkbf(float lo, float hi) {   // pack 2xf32 -> 2xbf16 dword
  v2bf p;
  p[0] = (__bf16)lo;
  p[1] = (__bf16)hi;
  return __builtin_bit_cast(unsigned int, p);   // -> v_cvt_pk_bf16_f32
}

DEVFN v8f wmma_bf16(v16bf a, v16bf b, v8f c) {
  return __builtin_amdgcn_wmma_f32_16x16x32_bf16(
      false, a, false, b, (short)0, c, false, false);
}

// Weight A-fragment (16x32 bf16), streamed from pre-packed global (L2 hot).
DEVFN v16bf load_frag(const u32x4* base, int frag, int lane) {
  Frag f;
  const u32x4* p = base + (unsigned)(frag * 32 + lane) * 2u;
  f.q[0] = p[0];
  f.q[1] = p[1];
  return f.v;
}

// Two consecutive 16-row C tiles (f32) -> one 32-row B fragment (bf16).
// B layout: reg r, lane l: n=l&15, k = (r<4?0:16) + (l>=16?8:0) + 2*(r&3)+{0,1}
// C layout: tile ct, reg r', : row = ct*16 + r' + 8*(l>=16), col = l&15
// => B reg r (r<4) = pack(c0[2r], c0[2r+1]); (r>=4) from c1. Pure in-lane.
DEVFN v16bf c2b(v8f c0, v8f c1) {
  Frag f;
#pragma unroll
  for (int r = 0; r < 4; ++r) {
    f.u[r]     = pkbf(c0[2 * r], c0[2 * r + 1]);
    f.u[r + 4] = pkbf(c1[2 * r], c1[2 * r + 1]);
  }
  return f.v;
}

DEVFN v8f splat8(float s) { v8f v = {s, s, s, s, s, s, s, s}; return v; }

DEVFN float relu1(float x) {                    // exact: wmma never makes sNaN
  float r;
  asm("v_max_num_f32 %0, 0, %1" : "=v"(r) : "v"(x));
  return r;
}

DEVFN v8f relu8(v8f c) {
#pragma unroll
  for (int r = 0; r < 8; ++r) c[r] = relu1(c[r]);
  return c;
}

// Bias C-fragment: value depends on row m = mt*16 + r + 8*lhalf only.
// 8 consecutive LDS floats -> compiler merges to ds_load_b128 x2.
DEVFN v8f bias_frag(const float* b, int mt, int lhalf) {
  const float* p = b + mt * 16 + 8 * lhalf;
  v8f acc;
#pragma unroll
  for (int r = 0; r < 8; ++r) acc[r] = p[r];
  return acc;
}

DEVFN float fast_tanh(float x) {
#if __has_builtin(__builtin_amdgcn_tanhf)
  return __builtin_amdgcn_tanhf(x);             // native v_tanh_f32
#elif __has_builtin(__builtin_amdgcn_tanh_f32)
  return __builtin_amdgcn_tanh_f32(x);
#else
  float ax = fabsf(x);
  float e  = __expf(-2.f * ax);
  float t  = (1.f - e) * __builtin_amdgcn_rcpf(1.f + e);
  return copysignf(t, x);
#endif
}

// vector_field on transposed state. yb0/yb1: state as B-fragments (K=64).
// dy[4]: output dy^T C-fragments (64 hidden rows x 16 batch cols).
DEVFN void vfield(v16bf yb0, v16bf yb1,
                  const u32x4* w0f, const u32x4* w1f, const u32x4* w2f,
                  const float* b0, const float* b1, const float* b2,
                  const float* cdv, int lane, v8f dy[4]) {
  int lhalf = lane >> 4;

  // ---- layer 1: H1^T(128x16) = W0(128x64) @ Y^T, relu, -> B frags ----
  v16bf h1b[4];
#pragma unroll
  for (int j = 0; j < 4; ++j) {
    v8f c0 = bias_frag(b0, 2 * j, lhalf);
    c0 = wmma_bf16(load_frag(w0f, (2 * j) * 2 + 0, lane), yb0, c0);
    c0 = wmma_bf16(load_frag(w0f, (2 * j) * 2 + 1, lane), yb1, c0);
    v8f c1 = bias_frag(b0, 2 * j + 1, lhalf);
    c1 = wmma_bf16(load_frag(w0f, (2 * j + 1) * 2 + 0, lane), yb0, c1);
    c1 = wmma_bf16(load_frag(w0f, (2 * j + 1) * 2 + 1, lane), yb1, c1);
    h1b[j] = c2b(relu8(c0), relu8(c1));
  }

  // ---- layer 2: H2^T(128x16) = W1(128x128) @ H1^T, relu, -> B frags ----
  v16bf h2b[4];
#pragma unroll
  for (int j = 0; j < 4; ++j) {
    v8f c0 = bias_frag(b1, 2 * j, lhalf);
#pragma unroll
    for (int kt = 0; kt < 4; ++kt)
      c0 = wmma_bf16(load_frag(w1f, (2 * j) * 4 + kt, lane), h1b[kt], c0);
    v8f c1 = bias_frag(b1, 2 * j + 1, lhalf);
#pragma unroll
    for (int kt = 0; kt < 4; ++kt)
      c1 = wmma_bf16(load_frag(w1f, (2 * j + 1) * 4 + kt, lane), h1b[kt], c1);
    h2b[j] = c2b(relu8(c0), relu8(c1));
  }

  // ---- layer 3: Z(1024x16) = W2(1024x128) @ H2^T; tanh; contract cd ----
  // Z tile zt covers rows m = zt*16 + (r + 8*lhalf); hid = zt, d = r+8*lhalf.
  float cdr[8];
#pragma unroll
  for (int r = 0; r < 8; ++r) cdr[r] = cdv[r + 8 * lhalf];
#pragma unroll
  for (int t = 0; t < 4; ++t) {
    v8f dacc = splat8(0.f);
    for (int mm = 0; mm < 16; ++mm) {          // rolled: bounds VGPR pressure
      int zt = t * 16 + mm;
      v8f acc = bias_frag(b2, zt, lhalf);
#pragma unroll
      for (int kt = 0; kt < 4; ++kt)
        acc = wmma_bf16(load_frag(w2f, zt * 4 + kt, lane), h2b[kt], acc);
      float s = 0.f;
#pragma unroll
      for (int r = 0; r < 8; ++r) s += fast_tanh(acc[r]) * cdr[r];
      s += __shfl_xor(s, 16);                  // combine the two d-halves
      int dr   = mm & 7;
      int bit  = mm >> 3;
      bool mine = (lhalf == bit);
#pragma unroll
      for (int r = 0; r < 8; ++r)
        dacc[r] = (mine && (r == dr)) ? s : dacc[r];
    }
    dy[t] = dacc;
  }
}

DEVFN void cpy(float* d, const float* s, int n, int tid, int nthr) {
  for (int i = tid; i < n; i += nthr) d[i] = s[i];
}

#define WAVES 4

__global__ __launch_bounds__(128) __attribute__((amdgpu_waves_per_eu(4)))
void cde_kernel(
    const float* __restrict__ ts, const float* __restrict__ x0,
    const float* __restrict__ lin1_w, const float* __restrict__ lin1_b,
    const float* __restrict__ b0g, const float* __restrict__ b1g,
    const float* __restrict__ b2g, const float* __restrict__ lin2_w,
    const float* __restrict__ lin2_b,
    const unsigned int* __restrict__ w0f_, const unsigned int* __restrict__ w1f_,
    const unsigned int* __restrict__ w2f_, const float* __restrict__ cdt,
    float* __restrict__ out) {
  __shared__ float s_lin1w[64 * 16];
  __shared__ float s_lin1b[64];
  __shared__ float s_lin2w[10 * 64];
  __shared__ float s_lin2b[10];
  __shared__ float s_b0[128];
  __shared__ float s_b1[128];
  __shared__ float s_b2[1024];
  __shared__ float s_cd[101 * 16];
  __shared__ float s_stage[WAVES][16 * 64];   // epilogue only

  int tid = threadIdx.x, nthr = blockDim.x;
  cpy(s_lin1w, lin1_w, 64 * 16, tid, nthr);
  cpy(s_lin1b, lin1_b, 64, tid, nthr);
  cpy(s_lin2w, lin2_w, 10 * 64, tid, nthr);
  cpy(s_lin2b, lin2_b, 10, tid, nthr);
  cpy(s_b0, b0g, 128, tid, nthr);
  cpy(s_b1, b1g, 128, tid, nthr);
  cpy(s_b2, b2g, 1024, tid, nthr);
  cpy(s_cd, cdt, 101 * 16, tid, nthr);
  __syncthreads();

  int wave = tid >> 5, lane = tid & 31;
  int cl = lane & 15, lhalf = lane >> 4;
  int m0 = (blockIdx.x * WAVES + wave) * 16;

  const u32x4* w0f = (const u32x4*)w0f_;
  const u32x4* w1f = (const u32x4*)w1f_;
  const u32x4* w2f = (const u32x4*)w2f_;

  float dt = (ts[200] - ts[0]) / 100.f;

  // ---- y0^T = lin1_w @ x0^T + lin1_b, directly into C-fragments ----
  float xr[16];                                 // this lane's batch row
#pragma unroll
  for (int d = 0; d < 16; ++d) xr[d] = x0[(m0 + cl) * 16 + d];
  v8f y[4];
#pragma unroll
  for (int t = 0; t < 4; ++t) {
#pragma unroll
    for (int r = 0; r < 8; ++r) {
      int m = t * 16 + r + 8 * lhalf;
      float s = s_lin1b[m];
#pragma unroll
      for (int d = 0; d < 16; ++d) s += s_lin1w[m * 16 + d] * xr[d];
      y[t][r] = s;
    }
  }

  // ---- 100 Heun steps ----
  v8f k[4];
#pragma unroll 1
  for (int step = 0; step < 100; ++step) {
    // Stop LICM from hoisting the (loop-invariant) weight/bias loads out of
    // the step loop -- it otherwise hoists then spills them to scratch.
    asm volatile("" ::: "memory");

    v16bf yb0 = c2b(y[0], y[1]);
    v16bf yb1 = c2b(y[2], y[3]);
    vfield(yb0, yb1, w0f, w1f, w2f, s_b0, s_b1, s_b2,
           s_cd + step * 16, lane, k);
    // z = y + dt*k1 (as bf16 B-frags), then fold 0.5*dt*k1 into y
    v8f z0 = y[0] + dt * k[0], z1 = y[1] + dt * k[1];
    v8f z2 = y[2] + dt * k[2], z3 = y[3] + dt * k[3];
    v16bf zb0 = c2b(z0, z1);
    v16bf zb1 = c2b(z2, z3);
#pragma unroll
    for (int t = 0; t < 4; ++t) y[t] = y[t] + (0.5f * dt) * k[t];
    vfield(zb0, zb1, w0f, w1f, w2f, s_b0, s_b1, s_b2,
           s_cd + (step + 1) * 16, lane, k);
#pragma unroll
    for (int t = 0; t < 4; ++t) y[t] = y[t] + (0.5f * dt) * k[t];
  }

  // ---- logits + softmax (after all WMMA; divergence is safe now) ----
  float* stf = s_stage[wave];                   // [batch col][hidden]
#pragma unroll
  for (int t = 0; t < 4; ++t)
#pragma unroll
    for (int r = 0; r < 8; ++r)
      stf[cl * 64 + t * 16 + r + 8 * lhalf] = y[t][r];

  if (lane < 16) {                              // one lane per batch row
    int row = m0 + lane;
    float lg[10];
#pragma unroll
    for (int j = 0; j < 10; ++j) {
      float s = s_lin2b[j];
      for (int h = 0; h < 64; ++h) s += stf[lane * 64 + h] * s_lin2w[j * 64 + h];
      lg[j] = s;
    }
    float mx = lg[0];
#pragma unroll
    for (int j = 1; j < 10; ++j) mx = fmaxf(mx, lg[j]);
    float se = 0.f;
#pragma unroll
    for (int j = 0; j < 10; ++j) { lg[j] = __expf(lg[j] - mx); se += lg[j]; }
    float inv = 1.f / se;
#pragma unroll
    for (int j = 0; j < 10; ++j) out[row * 10 + j] = lg[j] * inv;
  }
}

// ---- prep: pack (M,K) row-major f32 weight into bf16 WMMA A-fragments ----
// frag order [mtile][ktile]; per frag: 32 lanes x 8 dwords (2 bf16 each).
// A layout: reg r, lane l: m = l&15, k = kt*32+(l>=16?8:0)+(r<4?0:16)+2(r&3)+{0,1}
__global__ void pack_weights(const float* __restrict__ w,
                             unsigned int* __restrict__ dst, int K, int N) {
  int frag = blockIdx.x, lane = threadIdx.x & 31;
  int ktiles = K >> 5;
  int mt = frag / ktiles, kt = frag - mt * ktiles;
  int m  = mt * 16 + (lane & 15);
  int kb = kt * 32 + ((lane >> 4) << 3);
  unsigned int* o = dst + (unsigned)(frag * 32 + lane) * 8u;
#pragma unroll
  for (int r = 0; r < 8; ++r) {
    int k = kb + ((r & 3) << 1) + ((r >> 2) << 4);
    unsigned int lo = f2bf(w[m * K + k]);
    unsigned int hi = f2bf(w[m * K + k + 1]);
    o[r] = lo | (hi << 16);
  }
  (void)N;
}

// ---- prep: cd(t_i) table for i = 0..100 (replicates reference t accumulation
//      and searchsorted-left on ts[1:], cubic spline eval) ----
__global__ void build_cd(const float* __restrict__ ts,
                         const float* __restrict__ coeffs,
                         float* __restrict__ cdt) {
  int i = threadIdx.x + blockIdx.x * blockDim.x;
  if (i > 100) return;
  float dt = (ts[200] - ts[0]) / 100.f;
  float t = ts[0];
  for (int j = 0; j < i; ++j) t += dt;         // exact float accumulation
  int idx = 0;
  for (int j = 0; j < 200; ++j) idx += (ts[1 + j] < t) ? 1 : 0;
  if (idx > 199) idx = 199;
  float dl = t - ts[idx];
  for (int d = 0; d < 16; ++d) {
    const float* c = coeffs + (idx * 16 + d) * 4;
    cdt[i * 16 + d] = c[0] + dl * (c[1] + dl * (c[2] + dl * c[3]));
  }
}

extern "C" void kernel_launch(void* const* d_in, const int* in_sizes, int n_in,
                              void* d_out, int out_size, void* d_ws,
                              size_t ws_size, hipStream_t stream) {
  (void)in_sizes; (void)n_in; (void)out_size; (void)ws_size;
  const float* ts     = (const float*)d_in[0];
  const float* coeffs = (const float*)d_in[1];
  const float* x0     = (const float*)d_in[2];
  const float* lin1_w = (const float*)d_in[3];
  const float* lin1_b = (const float*)d_in[4];
  const float* mlp_w0 = (const float*)d_in[5];
  const float* mlp_b0 = (const float*)d_in[6];
  const float* mlp_w1 = (const float*)d_in[7];
  const float* mlp_b1 = (const float*)d_in[8];
  const float* mlp_w2 = (const float*)d_in[9];
  const float* mlp_b2 = (const float*)d_in[10];
  const float* lin2_w = (const float*)d_in[11];
  const float* lin2_b = (const float*)d_in[12];

  unsigned int* ws  = (unsigned int*)d_ws;
  unsigned int* w0f = ws;                        // 16 frags  * 256 dwords
  unsigned int* w1f = ws + 16 * 256;             // 32 frags
  unsigned int* w2f = ws + (16 + 32) * 256;      // 256 frags
  float*        cdt = (float*)(ws + (16 + 32 + 256) * 256);  // 101*16 f32

  pack_weights<<<16, 32, 0, stream>>>(mlp_w0, w0f, 64, 128);
  pack_weights<<<32, 32, 0, stream>>>(mlp_w1, w1f, 128, 128);
  pack_weights<<<256, 32, 0, stream>>>(mlp_w2, w2f, 128, 1024);
  build_cd<<<1, 128, 0, stream>>>(ts, coeffs, cdt);

  cde_kernel<<<64, 128, 0, stream>>>(ts, x0, lin1_w, lin1_b, mlp_b0, mlp_b1,
                                     mlp_b2, lin2_w, lin2_b, w0f, w1f, w2f,
                                     cdt, (float*)d_out);
}
